// AttentionRNNDecoder_hall_87428354277946
// MI455X (gfx1250) — matrix operand order repaired and checked by reading, not compiled, not run on hardware
//
#include <hip/hip_runtime.h>
#include <math.h>

#define BQ   64
#define ENCQ 256
#define DECQ 128
#define HQ   512

typedef __attribute__((ext_vector_type(16))) __bf16 v16bf;
typedef __attribute__((ext_vector_type(8)))  float  v8f;

union ABFrag { v16bf v; unsigned u[8]; };

__device__ __forceinline__ unsigned short f2bf(float f) {
  unsigned u = __float_as_uint(f);
  unsigned r = u + 0x7FFFu + ((u >> 16) & 1u);   // round-to-nearest-even
  return (unsigned short)(r >> 16);
}
__device__ __forceinline__ float bf2f(unsigned short h) {
  return __uint_as_float(((unsigned)h) << 16);
}
__device__ __forceinline__ float sigf(float x) {
  return 1.0f / (1.0f + __expf(-x));
}

// A-matrix 16x32 bf16 fragment (ISA 7.12.2): lane = M row (lane&15),
// half = lane>>4 selects K sub-ranges. VGPR v holds K pair:
//   k = (v>=4 ? 16 : 0) + half*8 + (v&3)*2
__device__ __forceinline__ v16bf load_afrag(const unsigned short* base, int lda,
                                            int m, int half, int kbase) {
  ABFrag r;
  const unsigned short* p = base + (size_t)m * lda + kbase;
#pragma unroll
  for (int v = 0; v < 8; ++v) {
    int k = ((v & 4) ? 16 : 0) + half * 8 + (v & 3) * 2;
    r.u[v] = *(const unsigned*)(p + k);
  }
  return r.v;
}

// B-matrix 32x16 bf16 fragment: lane = N col (lane&15); same K order as A:
//   VGPR v holds K pair k = (v>=4 ? 16 : 0) + half*8 + (v&3)*2
// Weights row-major [N, K]; streamed once per step -> non-temporal hint
// keeps WGP$ free for the small reused activation buffers.
__device__ __forceinline__ v16bf load_bfrag_nt(const unsigned short* base, int ldk,
                                               int n, int half, int kbase) {
  ABFrag r;
  const unsigned* p = (const unsigned*)(base + (size_t)n * ldk + kbase);
#pragma unroll
  for (int v = 0; v < 8; ++v) {
    r.u[v] = __builtin_nontemporal_load(p + ((v & 4) ? 8 : 0) + half * 4 + (v & 3));
  }
  return r.v;
}

// ---------------- setup kernels (once per call) ----------------

__global__ void __launch_bounds__(256) zero_ws_k(unsigned* p, int n) {
  for (int i = blockIdx.x * blockDim.x + threadIdx.x; i < n;
       i += gridDim.x * blockDim.x)
    p[i] = 0u;
}

__global__ void __launch_bounds__(256) conv_bf16_k(const float* __restrict__ in,
                                                   unsigned short* __restrict__ out,
                                                   int n) {
  for (int i = blockIdx.x * blockDim.x + threadIdx.x; i < n;
       i += gridDim.x * blockDim.x)
    out[i] = f2bf(in[i]);
}

// Wc[n, 0:ki] = W_ih[n, :] ; Wc[n, ki:ki+kh] = W_hh[n, :]  (bf16, [4H x (ki+kh)])
__global__ void __launch_bounds__(256) pack_w_k(const float* __restrict__ Wi,
                                                const float* __restrict__ Wh,
                                                unsigned short* __restrict__ out,
                                                int ki, int kh) {
  int K = ki + kh;
  int total = 4 * HQ * K;
  for (int idx = blockIdx.x * blockDim.x + threadIdx.x; idx < total;
       idx += gridDim.x * blockDim.x) {
    int n = idx / K, k = idx - n * K;
    float v = (k < ki) ? Wi[n * ki + k] : Wh[n * kh + (k - ki)];
    out[idx] = f2bf(v);
  }
}

__global__ void __launch_bounds__(256) bias_sum_k(const float* a0, const float* b0,
                                                  const float* a1, const float* b1,
                                                  float* o0, float* o1) {
  int i = blockIdx.x * blockDim.x + threadIdx.x;
  if (i < 4 * HQ) { o0[i] = a0[i] + b0[i]; o1[i] = a1[i] + b1[i]; }
}

// ---------------- per-step kernels ----------------

// scores[64,256] = Acat(bf16 [64x1536]) @ Wab^T + b_attn, masked.
// 64 tiles of 16x16, one wave each: grid 8 x 256.
__global__ void __launch_bounds__(256) scores_wmma_k(
    const unsigned short* __restrict__ Acat,
    const unsigned short* __restrict__ Wab,
    const float* __restrict__ b_attn,
    const int* __restrict__ x2_mask,
    float* __restrict__ scores) {
  int wave = blockIdx.x * 8 + (threadIdx.x >> 5);
  int lane = threadIdx.x & 31;
  int mtile = wave & 3, ntile = wave >> 2;
  int lidx = lane & 15, half = lane >> 4;
  int m = mtile * 16 + lidx, n = ntile * 16 + lidx;
  v8f acc = {};
  for (int kb = 0; kb < 48; ++kb) {
    v16bf a = load_afrag(Acat, 1536, m, half, kb * 32);
    v16bf b = load_bfrag_nt(Wab, 1536, n, half, kb * 32);
    acc = __builtin_amdgcn_wmma_f32_16x16x32_bf16(false, a, false, b,
                                                  (short)0, acc, false, false);
  }
  float bn = b_attn[n];
#pragma unroll
  for (int v = 0; v < 8; ++v) {
    int mr = mtile * 16 + v + 8 * half;
    float val = acc[v] + bn;
    if (x2_mask[mr * ENCQ + n]) val = -__builtin_inff();
    scores[mr * ENCQ + n] = val;
  }
}

// one workgroup per batch row: softmax over ENC, then ctx = attn @ enc[b]
// writes ctx (bf16) into A0cur[b, 0:1024]. Encoder rows are streamed as
// packed dwords (2 x bf16) with a non-temporal hint.
__global__ void __launch_bounds__(256) softmax_ctx_k(
    const float* __restrict__ scores,
    const unsigned short* __restrict__ encb,
    unsigned short* __restrict__ A0cur) {
  __shared__ float red[256];
  __shared__ float attn[256];
  int b = blockIdx.x, tid = threadIdx.x;
  float s = scores[b * ENCQ + tid];
  red[tid] = s; __syncthreads();
  for (int o = 128; o > 0; o >>= 1) {
    if (tid < o) red[tid] = fmaxf(red[tid], red[tid + o]);
    __syncthreads();
  }
  float mx = red[0]; __syncthreads();
  float e = __expf(s - mx);
  red[tid] = e; __syncthreads();
  for (int o = 128; o > 0; o >>= 1) {
    if (tid < o) red[tid] += red[tid + o];
    __syncthreads();
  }
  float inv = 1.0f / red[0];
  attn[tid] = e * inv;
  __syncthreads();
  // thread tid owns ctx columns {2*tid, 2*tid+1, 512+2*tid, 512+2*tid+1}
  float a00 = 0.f, a01 = 0.f, a10 = 0.f, a11 = 0.f;
  const unsigned short* eb = encb + (size_t)b * ENCQ * 1024;
  for (int ee = 0; ee < ENCQ; ++ee) {
    float w = attn[ee];
    const unsigned* rowd = (const unsigned*)(eb + ee * 1024);
    unsigned u0 = __builtin_nontemporal_load(rowd + tid);
    unsigned u1 = __builtin_nontemporal_load(rowd + tid + 256);
    a00 += w * bf2f((unsigned short)(u0 & 0xFFFFu));
    a01 += w * bf2f((unsigned short)(u0 >> 16));
    a10 += w * bf2f((unsigned short)(u1 & 0xFFFFu));
    a11 += w * bf2f((unsigned short)(u1 >> 16));
  }
  unsigned* orowd = (unsigned*)(A0cur + (size_t)b * 1536);
  orowd[tid]       = (unsigned)f2bf(a00) | ((unsigned)f2bf(a01) << 16);
  orowd[tid + 256] = (unsigned)f2bf(a10) | ((unsigned)f2bf(a11) << 16);
}

// Fused LSTM cell: each wave owns 16 batch rows x 16 hidden units and
// accumulates all 4 gates (i,f,g,o) -> local epilogue for c/h update.
// 128 wave-tasks: grid 16 x 256.
__global__ void __launch_bounds__(256) lstm_wmma_k(
    const unsigned short* __restrict__ A, int lda, int nkb,
    const unsigned short* __restrict__ W,
    const float* __restrict__ bias,
    float* __restrict__ cstate,
    unsigned short* __restrict__ d1, int s1,
    unsigned short* __restrict__ d2, int s2,
    unsigned short* __restrict__ d3, int s3) {
  int wave = blockIdx.x * 8 + (threadIdx.x >> 5);
  int lane = threadIdx.x & 31;
  int mtile = wave & 3, htile = wave >> 2;
  int lidx = lane & 15, half = lane >> 4;
  int m = mtile * 16 + lidx;
  int u = htile * 16 + lidx;
  v8f ai = {}, af = {}, ag = {}, ao = {};
  for (int kb = 0; kb < nkb; ++kb) {
    int kbase = kb * 32;
    v16bf a = load_afrag(A, lda, m, half, kbase);
    v16bf b0 = load_bfrag_nt(W, lda, 0 * HQ + u, half, kbase);
    ai = __builtin_amdgcn_wmma_f32_16x16x32_bf16(false, a, false, b0, (short)0, ai, false, false);
    v16bf b1 = load_bfrag_nt(W, lda, 1 * HQ + u, half, kbase);
    af = __builtin_amdgcn_wmma_f32_16x16x32_bf16(false, a, false, b1, (short)0, af, false, false);
    v16bf b2 = load_bfrag_nt(W, lda, 2 * HQ + u, half, kbase);
    ag = __builtin_amdgcn_wmma_f32_16x16x32_bf16(false, a, false, b2, (short)0, ag, false, false);
    v16bf b3 = load_bfrag_nt(W, lda, 3 * HQ + u, half, kbase);
    ao = __builtin_amdgcn_wmma_f32_16x16x32_bf16(false, a, false, b3, (short)0, ao, false, false);
  }
  float bi = bias[u], bfv = bias[HQ + u], bg = bias[2 * HQ + u], bo = bias[3 * HQ + u];
#pragma unroll
  for (int v = 0; v < 8; ++v) {
    int mr = mtile * 16 + v + 8 * half;
    float i_ = sigf(ai[v] + bi);
    float f_ = sigf(af[v] + bfv);
    float g_ = tanhf(ag[v] + bg);
    float o_ = sigf(ao[v] + bo);
    int ci = mr * HQ + u;
    float cn = f_ * cstate[ci] + i_ * g_;
    cstate[ci] = cn;
    unsigned short hb = f2bf(o_ * tanhf(cn));
    d1[mr * s1 + u] = hb;
    if (d2) d2[mr * s2 + u] = hb;
    if (d3) d3[mr * s3 + u] = hb;
  }
}

// out[64,1024] = h1(bf16) @ W_out^T + b_out ; write f32 to d_out[:,t,:] (NT)
// and bf16 back into Acat[:,0:1024]. 256 tiles: grid 32 x 256.
__global__ void __launch_bounds__(256) out_wmma_k(
    const unsigned short* __restrict__ Ah1,
    const unsigned short* __restrict__ Wob,
    const float* __restrict__ b_out,
    float* __restrict__ dout, int t,
    unsigned short* __restrict__ Acat) {
  int wave = blockIdx.x * 8 + (threadIdx.x >> 5);
  int lane = threadIdx.x & 31;
  int mtile = wave & 3, ntile = wave >> 2;  // ntile 0..63
  int lidx = lane & 15, half = lane >> 4;
  int m = mtile * 16 + lidx, n = ntile * 16 + lidx;
  v8f acc = {};
  for (int kb = 0; kb < 16; ++kb) {
    v16bf a = load_afrag(Ah1, 1024, m, half, kb * 32);
    v16bf b = load_bfrag_nt(Wob, 512, n, half, kb * 32);
    acc = __builtin_amdgcn_wmma_f32_16x16x32_bf16(false, a, false, b,
                                                  (short)0, acc, false, false);
  }
  float bn = b_out[n];
#pragma unroll
  for (int v = 0; v < 8; ++v) {
    int mr = mtile * 16 + v + 8 * half;
    float val = acc[v] + bn;
    __builtin_nontemporal_store(val, &dout[(size_t)mr * (DECQ * 1024) +
                                           (size_t)t * 1024 + n]);
    Acat[mr * 1536 + n] = f2bf(val);
  }
}

// ---------------- host-side orchestration ----------------

extern "C" void kernel_launch(void* const* d_in, const int* in_sizes, int n_in,
                              void* d_out, int out_size, void* d_ws, size_t ws_size,
                              hipStream_t stream) {
  (void)in_sizes; (void)n_in; (void)out_size; (void)ws_size;
  const float* enc    = (const float*)d_in[0];
  const int*   x2m    = (const int*)d_in[1];
  const float* W_ih0  = (const float*)d_in[2];
  const float* W_hh0  = (const float*)d_in[3];
  const float* b_ih0  = (const float*)d_in[4];
  const float* b_hh0  = (const float*)d_in[5];
  const float* W_ih1  = (const float*)d_in[6];
  const float* W_hh1  = (const float*)d_in[7];
  const float* b_ih1  = (const float*)d_in[8];
  const float* b_hh1  = (const float*)d_in[9];
  const float* W_attn = (const float*)d_in[10];
  const float* b_attn = (const float*)d_in[11];
  const float* W_out  = (const float*)d_in[12];
  const float* b_out  = (const float*)d_in[13];
  float* out = (float*)d_out;

  char* ws = (char*)d_ws;
  size_t off = 0;
  auto alloc = [&](size_t bytes) -> char* {
    char* p = ws + off;
    off += (bytes + 255) & ~(size_t)255;
    return p;
  };
  unsigned short* encb = (unsigned short*)alloc(16777216ULL * 2);  // enc bf16
  unsigned short* Wc0  = (unsigned short*)alloc(3145728ULL * 2);   // [2048x1536]
  unsigned short* Wc1  = (unsigned short*)alloc(2097152ULL * 2);   // [2048x1024]
  unsigned short* Wab  = (unsigned short*)alloc(393216ULL * 2);    // [256x1536]
  unsigned short* Wob  = (unsigned short*)alloc(524288ULL * 2);    // [1024x512]
  float* bc0     = (float*)alloc(2048 * 4);
  float* bc1     = (float*)alloc(2048 * 4);
  float* scoresb = (float*)alloc(16384 * 4);
  size_t zs = off;                                   // zero-init region start
  unsigned short* Acat = (unsigned short*)alloc(98304ULL * 2);      // [64x1536]
  unsigned short* A0   = (unsigned short*)alloc(2 * 98304ULL * 2);  // 2x[64x1536]
  unsigned short* A1   = (unsigned short*)alloc(2 * 65536ULL * 2);  // 2x[64x1024]
  float* c0s = (float*)alloc(32768 * 4);
  float* c1s = (float*)alloc(32768 * 4);
  size_t zbytes = off - zs;

  zero_ws_k<<<512, 256, 0, stream>>>((unsigned*)(ws + zs), (int)(zbytes / 4));
  conv_bf16_k<<<4096, 256, 0, stream>>>(enc, encb, 16777216);
  conv_bf16_k<<<512, 256, 0, stream>>>(W_attn, Wab, 393216);
  conv_bf16_k<<<512, 256, 0, stream>>>(W_out, Wob, 524288);
  pack_w_k<<<2048, 256, 0, stream>>>(W_ih0, W_hh0, Wc0, 1024, 512);
  pack_w_k<<<2048, 256, 0, stream>>>(W_ih1, W_hh1, Wc1, 512, 512);
  bias_sum_k<<<8, 256, 0, stream>>>(b_ih0, b_hh0, b_ih1, b_hh1, bc0, bc1);

  for (int t = 0; t < DECQ; ++t) {
    int p = t & 1, np = p ^ 1;
    unsigned short* A0c = A0 + (size_t)p * 98304;
    unsigned short* A0n = A0 + (size_t)np * 98304;
    unsigned short* A1c = A1 + (size_t)p * 65536;
    unsigned short* A1n = A1 + (size_t)np * 65536;

    scores_wmma_k<<<8, 256, 0, stream>>>(Acat, Wab, b_attn, x2m, scoresb);
    softmax_ctx_k<<<64, 256, 0, stream>>>(scoresb, encb, A0c);
    // LSTM0: A=[ctx|h0_{t-1}] -> h0_t to: A0next h-part, Acat h-part, A1cur h0-part
    lstm_wmma_k<<<16, 256, 0, stream>>>(A0c, 1536, 48, Wc0, bc0, c0s,
                                        A0n + 1024, 1536,
                                        Acat + 1024, 1536,
                                        A1c, 1024);
    // LSTM1: A=[h0_t|h1_{t-1}] -> h1_t to A1next h1-part
    lstm_wmma_k<<<16, 256, 0, stream>>>(A1c, 1024, 32, Wc1, bc1, c1s,
                                        A1n + 512, 1024,
                                        nullptr, 0, nullptr, 0);
    out_wmma_k<<<32, 256, 0, stream>>>(A1n + 512, Wob, b_out, out, t, Acat);
  }
}
